// Strategist_84112639525122
// MI455X (gfx1250) — matrix operand order repaired and checked by reading, not compiled, not run on hardware
//
#include <hip/hip_runtime.h>

// ---------------------------------------------------------------------------
// BitNet-1.58 MLP forward for MI455X (gfx1250, wave32, WMMA bf16 16x16x32).
//   h = concat(x,y,s)@Wq_in^T + b_in
//   L x { u = gelu(h@Wq1^T+b1); h = h + u@Wq2^T+b2 }
//   z = h@Wq_z^T+b_z ; mask = sigmoid(h@Wq_m^T+b_m)
// GEMM: 128x128 block tile, 8 waves, 64x32 register tile per wave
// (8 WMMAs / K-step / wave), A+B staged in LDS with double-buffered
// global_load_async_to_lds_b128 (ASYNCcnt) DMA.  Workspace ~340 MB.
// ---------------------------------------------------------------------------

typedef __attribute__((ext_vector_type(16))) __bf16 v16bf;
typedef __attribute__((ext_vector_type(8)))  __bf16 v8bf;
typedef __attribute__((ext_vector_type(8)))  float  v8f;

static constexpr size_t NROWS = 256ull * 81;   // 20736 rows = 128 * 162
static constexpr int    DIM   = 1024;
static constexpr int    HID   = 4096;
static constexpr int    K_IN  = 3072;

// ---- workspace layout (bytes) ----
static constexpr size_t SC_OFF   = 0;                          // 7 scales
static constexpr size_t PART_OFF = 256;                        // 7*512 partials
static constexpr size_t WQIN_OFF = 16384;
static constexpr size_t WQIN_SZ  = 1024ull * 3072 * 2;
static constexpr size_t WQ1_OFF  = WQIN_OFF + WQIN_SZ;
static constexpr size_t WQ1_SZ   = 2ull * 4096 * 1024 * 2;
static constexpr size_t WQ2_OFF  = WQ1_OFF + WQ1_SZ;
static constexpr size_t WQ2_SZ   = 2ull * 1024 * 4096 * 2;
static constexpr size_t WQZ_OFF  = WQ2_OFF + WQ2_SZ;
static constexpr size_t WQZ_SZ   = 1024ull * 1024 * 2;
static constexpr size_t WQM_OFF  = WQZ_OFF + WQZ_SZ;
static constexpr size_t WQM_SZ   = 2048;
static constexpr size_t HF_OFF   = WQM_OFF + WQM_SZ;           // h (f32)
static constexpr size_t HF_SZ    = NROWS * 1024 * 4;
static constexpr size_t HB_OFF   = HF_OFF + HF_SZ;             // h (bf16)
static constexpr size_t HB_SZ    = NROWS * 1024 * 2;
static constexpr size_t XU_OFF   = HB_OFF + HB_SZ;             // X aliased w/ u

__device__ __forceinline__ float bf2f(__bf16 x) {
  unsigned short u = __builtin_bit_cast(unsigned short, x);
  unsigned int   v = ((unsigned int)u) << 16;
  return __builtin_bit_cast(float, v);
}
__device__ __forceinline__ __bf16 f2bf(float f) {
  unsigned int u = __builtin_bit_cast(unsigned int, f);
  unsigned int r = u + 0x7FFFu + ((u >> 16) & 1u);   // round-to-nearest-even
  unsigned short h = (unsigned short)(r >> 16);
  return __builtin_bit_cast(__bf16, h);
}

// ---- CDNA5 async DMA: global -> LDS, 16B per lane, tracked by ASYNCcnt ----
// lds_base is passed as an operand so the shared array ESCAPES into the asm:
// without it, LLVM proves the asm cannot write the (non-escaped) LDS alloc
// and folds all subsequent ds_load fragments to undef.
__device__ __forceinline__ void async_load_b128(unsigned lds_byte_off,
                                                const void* gaddr,
                                                void* lds_base) {
  unsigned long long ga = (unsigned long long)gaddr;
  asm volatile("global_load_async_to_lds_b128 %0, %1, off"
               :: "v"(lds_byte_off), "v"(ga), "v"(lds_base)
               : "memory");
}
__device__ __forceinline__ void wait_async0() {
  asm volatile("s_wait_asynccnt 0x0" ::: "memory");
}

// ---------------- deterministic |W| mean (two-stage, fixed order) ----------
__global__ __launch_bounds__(256) void absum_partial(const float* __restrict__ w,
                                                     long n, float* __restrict__ partials) {
  __shared__ float red[256];
  long stride = (long)gridDim.x * blockDim.x;
  float s = 0.f;
  for (long i = (long)blockIdx.x * blockDim.x + threadIdx.x; i < n; i += stride)
    s += fabsf(w[i]);
  red[threadIdx.x] = s;
  __syncthreads();
  for (int off = 128; off > 0; off >>= 1) {
    if ((int)threadIdx.x < off) red[threadIdx.x] += red[threadIdx.x + off];
    __syncthreads();
  }
  if (threadIdx.x == 0) partials[blockIdx.x] = red[0];
}

__global__ __launch_bounds__(512) void absum_final(const float* __restrict__ partials,
                                                   float* __restrict__ scale, float count) {
  __shared__ float red[512];
  red[threadIdx.x] = partials[threadIdx.x];
  __syncthreads();
  for (int off = 256; off > 0; off >>= 1) {
    if ((int)threadIdx.x < off) red[threadIdx.x] += red[threadIdx.x + off];
    __syncthreads();
  }
  if (threadIdx.x == 0) scale[0] = red[0] / count + 1e-8f;
}

// ---------------- ternary quantize -> bf16 ---------------------------------
__global__ __launch_bounds__(256) void quant_kernel(const float* __restrict__ w,
                                                    const float* __restrict__ scale,
                                                    __bf16* __restrict__ out, long n) {
  long i = (long)blockIdx.x * blockDim.x + threadIdx.x;
  if (i >= n) return;
  float a = scale[0];
  float q = fminf(fmaxf(rintf(w[i] / a), -1.f), 1.f);
  out[i] = f2bf(a * q);
}

// ---------------- pack concat(x,y,s) -> bf16 [N,3072] ----------------------
__global__ __launch_bounds__(256) void pack_x(const float* __restrict__ x,
                                              const float* __restrict__ y,
                                              const float* __restrict__ s,
                                              __bf16* __restrict__ X, long total) {
  long i = (long)blockIdx.x * blockDim.x + threadIdx.x;
  if (i >= total) return;
  int  c = (int)(i % K_IN);
  long n = i / K_IN;
  float v;
  if (c < DIM)            v = x[n * DIM + c];
  else if (c < 2 * DIM)   v = y[n * DIM + (c - DIM)];
  else                    v = s[n * DIM + (c - 2 * DIM)];
  X[i] = f2bf(v);
}

// ---------------- WMMA bf16 GEMM: C[N,OUT] = A[N,K] @ W[OUT,K]^T -----------
// 256 threads = 8 waves; block tile 128x128; wave tile 64x32 (4x2 wmma accs).
// A(128x32) and B(128x32) K-slabs double-buffered in LDS (32 KB), filled by
// global_load_async_to_lds_b128, synced with s_wait_asynccnt + barrier.
__global__ __launch_bounds__(256) void gemm_bf16_wmma(
    const __bf16* __restrict__ A, const __bf16* __restrict__ W,
    const float* __restrict__ bias, const float* __restrict__ residual,
    float* __restrict__ outF, __bf16* __restrict__ outB,
    int K, int OUT, int applyGelu) {
  // LDS: buf0 A[0,8K) B[8K,16K) | buf1 A[16K,24K) B[24K,32K)
  __shared__ char smem[32768];
  void* smem_escape = (void*)smem;

  const int tid   = threadIdx.x;
  const int lane  = tid & 31;
  const int wave  = tid >> 5;
  const int waveM = wave >> 2;                // 0..1 -> 64-row group
  const int waveN = wave & 3;                 // 0..3 -> 32-col group
  const int r16   = lane & 15;
  const int hi    = lane >> 4;
  const int m0 = blockIdx.x * 128;
  const int n0 = blockIdx.y * 128;

  // per-thread async slice: thread covers 16 bf16 of A tile + 16 of B tile
  const int aRow = tid >> 1;                  // 0..127
  const int aCol = (tid & 1) * 16;            // 0 or 16
  const size_t aOff = (size_t)(m0 + aRow) * K + aCol;
  const size_t bOff = (size_t)(n0 + aRow) * K + aCol;
  const unsigned tLds = (unsigned)(aRow * 64 + aCol * 2);  // bytes in tile

  v8f acc[4][2] = {};

  // prologue: fill buffer 0 with K-slab 0
  async_load_b128(tLds,             A + aOff,     smem_escape);
  async_load_b128(tLds + 16,        A + aOff + 8, smem_escape);
  async_load_b128(8192 + tLds,      W + bOff,     smem_escape);
  async_load_b128(8192 + tLds + 16, W + bOff + 8, smem_escape);

  const int nkb = K >> 5;
  for (int i = 0; i < nkb; ++i) {
    const unsigned base = (i & 1) ? 16384u : 0u;
    wait_async0();
    __syncthreads();                          // this K-slab visible to all waves

    if (i + 1 < nkb) {                        // prefetch next slab, other buffer
      const unsigned nb = ((i + 1) & 1) ? 16384u : 0u;
      const int kn = (i + 1) << 5;
      async_load_b128(nb + tLds,             A + aOff + kn,     smem_escape);
      async_load_b128(nb + tLds + 16,        A + aOff + kn + 8, smem_escape);
      async_load_b128(nb + 8192 + tLds,      W + bOff + kn,     smem_escape);
      async_load_b128(nb + 8192 + tLds + 16, W + bOff + kn + 8, smem_escape);
    }

    const __bf16* As = (const __bf16*)(smem + base);
    const __bf16* Bs = (const __bf16*)(smem + base + 8192);

    // A frags: lanes 0-15 hold K 0..7 & 16..23, lanes 16-31 hold 8..15 & 24..31
    v16bf a[4];
#pragma unroll
    for (int mt = 0; mt < 4; ++mt) {
      const int row = waveM * 64 + mt * 16 + r16;
      v8bf lo = *(const v8bf*)(As + row * 32 + hi * 8);
      v8bf hh = *(const v8bf*)(As + row * 32 + 16 + hi * 8);
      a[mt] = __builtin_shufflevector(lo, hh, 0, 1, 2, 3, 4, 5, 6, 7,
                                      8, 9, 10, 11, 12, 13, 14, 15);
    }
    // B frags: lane = out column; lanes 0-15 K 0..7 & 16..23, lanes 16-31 K 8..15 & 24..31
    v16bf b[2];
#pragma unroll
    for (int nt = 0; nt < 2; ++nt) {
      const int col = waveN * 32 + nt * 16 + r16;
      v8bf blo = *(const v8bf*)(Bs + col * 32 + hi * 8);
      v8bf bhh = *(const v8bf*)(Bs + col * 32 + 16 + hi * 8);
      b[nt] = __builtin_shufflevector(blo, bhh, 0, 1, 2, 3, 4, 5, 6, 7,
                                      8, 9, 10, 11, 12, 13, 14, 15);
    }

#pragma unroll
    for (int mt = 0; mt < 4; ++mt)
#pragma unroll
      for (int nt = 0; nt < 2; ++nt)
        acc[mt][nt] = __builtin_amdgcn_wmma_f32_16x16x32_bf16(
            false, a[mt], false, b[nt], (short)0, acc[mt][nt], false, false);

    __syncthreads();                          // all reads done before refill
  }

  // epilogue: bias / gelu / residual / f32+bf16 stores
#pragma unroll
  for (int mt = 0; mt < 4; ++mt) {
    const int mb = m0 + waveM * 64 + mt * 16 + hi * 8;  // C: VGPRe -> M=e/e+8
#pragma unroll
    for (int nt = 0; nt < 2; ++nt) {
      const int n = n0 + waveN * 32 + nt * 16 + r16;
      const float bi = bias[n];
#pragma unroll
      for (int e = 0; e < 8; ++e) {
        float v = acc[mt][nt][e] + bi;
        if (applyGelu) v = 0.5f * v * (1.0f + erff(v * 0.70710678118654752f));
        size_t o = (size_t)(mb + e) * OUT + n;
        if (residual) v += residual[o];
        if (outF) outF[o] = v;
        if (outB) outB[o] = f2bf(v);
      }
    }
  }
}

// ---------------- focus mask head (OUT=1) + sigmoid ------------------------
__global__ __launch_bounds__(128) void mask_kernel(const __bf16* __restrict__ h,
                                                   const __bf16* __restrict__ wm,
                                                   const float* __restrict__ bm,
                                                   float* __restrict__ out, int K) {
  __shared__ float red[128];
  const int row = blockIdx.x;
  const __bf16* hr = h + (size_t)row * K;
  float s = 0.f;
  for (int i = threadIdx.x; i < K; i += blockDim.x)
    s += bf2f(hr[i]) * bf2f(wm[i]);
  red[threadIdx.x] = s;
  __syncthreads();
  for (int off = 64; off > 0; off >>= 1) {
    if ((int)threadIdx.x < off) red[threadIdx.x] += red[threadIdx.x + off];
    __syncthreads();
  }
  if (threadIdx.x == 0) {
    float v = red[0] + bm[0];
    out[row] = 1.f / (1.f + expf(-v));
  }
}

// ---------------------------------------------------------------------------
extern "C" void kernel_launch(void* const* d_in, const int* in_sizes, int n_in,
                              void* d_out, int out_size, void* d_ws, size_t ws_size,
                              hipStream_t stream) {
  (void)in_sizes; (void)n_in; (void)out_size; (void)ws_size;
  const float* x   = (const float*)d_in[0];
  const float* y   = (const float*)d_in[1];
  const float* sp  = (const float*)d_in[2];
  const float* Win = (const float*)d_in[3];
  const float* bin = (const float*)d_in[4];
  const float* W1  = (const float*)d_in[5];
  const float* b1  = (const float*)d_in[6];
  const float* W2  = (const float*)d_in[7];
  const float* b2  = (const float*)d_in[8];
  const float* Wz  = (const float*)d_in[9];
  const float* bz  = (const float*)d_in[10];
  const float* Wm  = (const float*)d_in[11];
  const float* bm  = (const float*)d_in[12];
  float* out = (float*)d_out;

  char* ws = (char*)d_ws;
  float*  scales   = (float*)(ws + SC_OFF);
  float*  partials = (float*)(ws + PART_OFF);
  __bf16* wqin = (__bf16*)(ws + WQIN_OFF);
  __bf16* wq1  = (__bf16*)(ws + WQ1_OFF);
  __bf16* wq2  = (__bf16*)(ws + WQ2_OFF);
  __bf16* wqz  = (__bf16*)(ws + WQZ_OFF);
  __bf16* wqm  = (__bf16*)(ws + WQM_OFF);
  float*  hF   = (float*)(ws + HF_OFF);
  __bf16* hB   = (__bf16*)(ws + HB_OFF);
  __bf16* X    = (__bf16*)(ws + XU_OFF);
  __bf16* U    = (__bf16*)(ws + XU_OFF);   // u aliases X (X dead after proj_in)

  // --- per-tensor scales (7 tensors), deterministic two-stage reduction ---
  struct Tens { const float* w; long n; __bf16* dst; };
  const Tens tens[7] = {
      {Win,                   1024L * 3072, wqin},
      {W1,                    4096L * 1024, wq1},
      {W1 + 4096L * 1024,     4096L * 1024, wq1 + 4096L * 1024},
      {W2,                    1024L * 4096, wq2},
      {W2 + 1024L * 4096,     1024L * 4096, wq2 + 1024L * 4096},
      {Wz,                    1024L * 1024, wqz},
      {Wm,                    1024L,        wqm},
  };
  for (int t = 0; t < 7; ++t) {
    absum_partial<<<512, 256, 0, stream>>>(tens[t].w, tens[t].n, partials + t * 512);
    absum_final<<<1, 512, 0, stream>>>(partials + t * 512, scales + t, (float)tens[t].n);
  }
  for (int t = 0; t < 7; ++t) {
    long n = tens[t].n;
    quant_kernel<<<(int)((n + 255) / 256), 256, 0, stream>>>(tens[t].w, scales + t,
                                                             tens[t].dst, n);
  }

  // --- pack activations to bf16 [N, 3072] ---
  {
    long total = (long)NROWS * K_IN;
    pack_x<<<(int)((total + 255) / 256), 256, 0, stream>>>(x, y, sp, X, total);
  }

  const int gm = (int)(NROWS / 128);   // 162 row tiles

  // --- h = X @ Wq_in^T + b_in  (f32 + bf16 copies) ---
  gemm_bf16_wmma<<<dim3(gm, DIM / 128), 256, 0, stream>>>(
      X, wqin, bin, nullptr, hF, hB, K_IN, DIM, 0);

  // --- residual BitMLP layers ---
  for (int l = 0; l < 2; ++l) {
    // u = gelu(h @ Wq1[l]^T + b1[l])
    gemm_bf16_wmma<<<dim3(gm, HID / 128), 256, 0, stream>>>(
        hB, wq1 + (size_t)l * HID * DIM, b1 + (size_t)l * HID,
        nullptr, nullptr, U, DIM, HID, 1);
    // h = h + u @ Wq2[l]^T + b2[l]
    gemm_bf16_wmma<<<dim3(gm, DIM / 128), 256, 0, stream>>>(
        U, wq2 + (size_t)l * DIM * HID, b2 + (size_t)l * DIM,
        hF, hF, hB, HID, DIM, 0);
  }

  // --- z = h @ Wq_z^T + b_z  -> d_out[0 : N*DIM) ---
  gemm_bf16_wmma<<<dim3(gm, DIM / 128), 256, 0, stream>>>(
      hB, wqz, bz, nullptr, out, nullptr, DIM, DIM, 0);

  // --- focus_mask = sigmoid(h @ Wq_m^T + b_m) -> d_out[N*DIM : N*DIM+N) ---
  mask_kernel<<<(int)NROWS, 128, 0, stream>>>(hB, wqm, bm, out + NROWS * (size_t)DIM,
                                              DIM);
}
